// GrammarModule_24885040513455
// MI455X (gfx1250) — matrix operand order repaired and checked by reading, not compiled, not run on hardware
//
#include <hip/hip_runtime.h>

// ---------------------------------------------------------------------------
// GrammarModule forward for MI455X (gfx1250, wave32).
//  - all GEMMs: v_wmma_f32_16x16x32_bf16, f32 accumulate
//  - k_gemm: double-buffered async global->LDS staging
//    (global_load_async_to_lds_b128 + s_wait_asynccnt), bf16 operands,
//    weights pre-transposed so every copy is a contiguous 16-B segment
//  - fragment reads are ds_load_b128 via fragment-friendly LDS strides
// ---------------------------------------------------------------------------

typedef __attribute__((ext_vector_type(16))) __bf16 v16bf;
typedef __attribute__((ext_vector_type(8)))  __bf16 v8bf;
typedef __attribute__((ext_vector_type(8)))  float  v8f;

#define HID 512
#define DH 64
#define NHEAD 8
#define NSEQ 512
#define NBATCH 16
#define MROWS (NBATCH * NSEQ)   // 8192

__device__ __forceinline__ __bf16 f2bf(float f) {
  unsigned u = __builtin_bit_cast(unsigned, f);
  unsigned r = u + 0x7FFFu + ((u >> 16) & 1u);          // round-to-nearest-even
  unsigned short s = (unsigned short)(r >> 16);
  return __builtin_bit_cast(__bf16, s);
}

// async 16-byte global -> LDS copy (per-lane addresses), ASYNCcnt-tracked
__device__ __forceinline__ void async_b128(unsigned lds_off, const __bf16* g) {
  asm volatile("global_load_async_to_lds_b128 %0, %1, off"
               :: "v"(lds_off), "v"(g) : "memory");
}
__device__ __forceinline__ void wait_async0() {
  asm volatile("s_wait_asynccnt 0x0" ::: "memory");
}

// A-matrix 16x32 bf16 fragment from a row-major tile.
// Lane L (m=L&15, h=L>>4) holds K = {8h..8h+7} U {16+8h..16+8h+7}:
// two contiguous 16-B runs -> two ds_load_b128. ld multiple of 8 elements.
__device__ __forceinline__ v16bf load_frag_a(const __bf16* src, int ld) {
  int lane = threadIdx.x & 31;
  int m = lane & 15, hh = lane >> 4;
  const __bf16* p0 = src + m * ld + (hh << 3);
  v8bf lo = *(const v8bf*)(p0);
  v8bf hi = *(const v8bf*)(p0 + 16);
  v16bf a;
#pragma unroll
  for (int i = 0; i < 8; ++i) { a[i] = lo[i]; a[i + 8] = hi[i]; }
  return a;
}

// B-matrix 32x16 bf16 fragment from a TRANSPOSED tile Bt[n][k] (n-major).
// Lane L (n=L&15, h=L>>4) holds K = {8h..8h+7} U {16+8h..16+8h+7}.
__device__ __forceinline__ v16bf load_frag_bt(const __bf16* src, int ld) {
  int lane = threadIdx.x & 31;
  int n = lane & 15, hh = lane >> 4;
  const __bf16* p0 = src + n * ld + (hh << 3);
  v8bf lo = *(const v8bf*)(p0);
  v8bf hi = *(const v8bf*)(p0 + 16);
  v16bf b;
#pragma unroll
  for (int i = 0; i < 8; ++i) { b[i] = lo[i]; b[i + 8] = hi[i]; }
  return b;
}

// ---------------------------------------------------------------------------
// GEMM:  out = act(A[M,K](bf16) @ Wt[N,K]^T(bf16) + bias) [+ R]
// act: 0 none, 1 exact GELU. Writes f32 C or bf16 Cb (exactly one non-null).
// 64x128 tile, 8 waves, K chunks of 32, double-buffered async staging.
// A-tile rows and Wt-tile rows are copied as contiguous 16-B segments; OOB
// rows are clamped (their products land in C rows the epilogue never stores).
// ---------------------------------------------------------------------------
#define TM 64
#define TN 128
#define TK 32
#define LDA 40   // 80-B row stride: 16-B aligned, bank-conflict free

__global__ __launch_bounds__(256) void k_gemm(
    const __bf16* __restrict__ A, const __bf16* __restrict__ Wt,
    const float* __restrict__ bias, const float* __restrict__ R,
    float* __restrict__ C, __bf16* __restrict__ Cb,
    int M, int N, int K, int act)
{
  __shared__ __attribute__((aligned(16))) __bf16 sA [2][TM][LDA];
  __shared__ __attribute__((aligned(16))) __bf16 sBt[2][TN][LDA];

  int tid = threadIdx.x;
  int wave = tid >> 5;
  int row0 = blockIdx.y * TM;
  int col0 = blockIdx.x * TN;
  int wr = (wave & 3) << 4;   // wave row offset (0,16,32,48)
  int wc = (wave >> 2) << 6;  // wave col offset (0,64)

  // per-thread staging slots
  int ar = tid >> 2;                    // A: row 0..63
  int aseg = (tid & 3) << 3;            // A: 8-elem segment
  int agr = row0 + ar; if (agr >= M) agr = M - 1;          // clamp
  int bn0 = tid >> 2,  bseg0 = (tid & 3) << 3;             // Bt: rows 0..63
  int bn1 = 64 + bn0,  bseg1 = bseg0;                      // Bt: rows 64..127

  v8f acc[4] = {};

  // prologue: stage first K-chunk into buffer 0
  async_b128((unsigned)(uintptr_t)&sA [0][ar][aseg],  A  + (long)agr * K + aseg);
  async_b128((unsigned)(uintptr_t)&sBt[0][bn0][bseg0], Wt + (long)(col0 + bn0) * K + bseg0);
  async_b128((unsigned)(uintptr_t)&sBt[0][bn1][bseg1], Wt + (long)(col0 + bn1) * K + bseg1);

  int cur = 0;
  for (int k0 = 0; k0 < K; k0 += TK) {
    wait_async0();
    __syncthreads();
    if (k0 + TK < K) {
      int nb = cur ^ 1, kn = k0 + TK;
      async_b128((unsigned)(uintptr_t)&sA [nb][ar][aseg],  A  + (long)agr * K + kn + aseg);
      async_b128((unsigned)(uintptr_t)&sBt[nb][bn0][bseg0], Wt + (long)(col0 + bn0) * K + kn + bseg0);
      async_b128((unsigned)(uintptr_t)&sBt[nb][bn1][bseg1], Wt + (long)(col0 + bn1) * K + kn + bseg1);
    }
    v16bf a = load_frag_a(&sA[cur][wr][0], LDA);
#pragma unroll
    for (int t = 0; t < 4; ++t) {
      v16bf b = load_frag_bt(&sBt[cur][wc + t * 16][0], LDA);
      acc[t] = __builtin_amdgcn_wmma_f32_16x16x32_bf16(
          false, a, false, b, (short)0, acc[t], false, false);
    }
    cur ^= 1;
  }

  int lane = tid & 31;
  int nl = lane & 15, hh = lane >> 4;
#pragma unroll
  for (int t = 0; t < 4; ++t) {
    int col = col0 + wc + t * 16 + nl;
    if (col >= N) continue;
#pragma unroll
    for (int r = 0; r < 8; ++r) {
      int row = row0 + wr + (hh << 3) + r;
      if (row >= M) continue;
      float v = acc[t][r];
      if (bias) v += bias[col];
      if (act == 1) v = 0.5f * v * (1.0f + erff(v * 0.70710678118654752f));
      if (R) v += R[(long)row * N + col];
      if (Cb) Cb[(long)row * N + col] = f2bf(v);
      else    C [(long)row * N + col] = v;
    }
  }
}

// ---------------------------------------------------------------------------
// f32 -> bf16 conversion (n multiple of 4)
// ---------------------------------------------------------------------------
__global__ __launch_bounds__(256) void k_cvt(
    const float* __restrict__ src, __bf16* __restrict__ dst, long n)
{
  long i = ((long)blockIdx.x * 256 + threadIdx.x) * 4;
  if (i + 3 < n) {
    float4 v = *(const float4*)&src[i];
    dst[i]     = f2bf(v.x); dst[i + 1] = f2bf(v.y);
    dst[i + 2] = f2bf(v.z); dst[i + 3] = f2bf(v.w);
  }
}

// ---------------------------------------------------------------------------
// Weight transpose + convert: Wt[n*K+k] = bf16(W[k*N+n])
// ---------------------------------------------------------------------------
__global__ __launch_bounds__(256) void k_wtr(
    const float* __restrict__ W, __bf16* __restrict__ Wt, int K, int N)
{
  long idx = (long)blockIdx.x * 256 + threadIdx.x;
  if (idx >= (long)K * N) return;
  int n = (int)(idx / K), k = (int)(idx % K);
  Wt[idx] = f2bf(W[(long)k * N + n]);
}

// ---------------------------------------------------------------------------
// LayerNorm over last dim D; writes bf16 (GEMM A-operand). One block per row.
// ---------------------------------------------------------------------------
__global__ __launch_bounds__(256) void k_layernorm(
    const float* __restrict__ X, const float* __restrict__ g,
    const float* __restrict__ bta, __bf16* __restrict__ Y, int D)
{
  __shared__ float red[256];
  long row = blockIdx.x;
  const float* x = X + row * D;
  int tid = threadIdx.x;

  float s = 0.0f;
  for (int c = tid; c < D; c += 256) s += x[c];
  red[tid] = s; __syncthreads();
  for (int st = 128; st > 0; st >>= 1) {
    if (tid < st) red[tid] += red[tid + st];
    __syncthreads();
  }
  float mu = red[0] / (float)D;
  __syncthreads();

  float vs = 0.0f;
  for (int c = tid; c < D; c += 256) { float d = x[c] - mu; vs += d * d; }
  red[tid] = vs; __syncthreads();
  for (int st = 128; st > 0; st >>= 1) {
    if (tid < st) red[tid] += red[tid + st];
    __syncthreads();
  }
  float inv = rsqrtf(red[0] / (float)D + 1e-5f);
  for (int c = tid; c < D; c += 256)
    Y[row * D + c] = f2bf((x[c] - mu) * inv * g[c] + bta[c]);
}

// ---------------------------------------------------------------------------
// x += pe[posc] + pos_tab[posc] + strand_tab[strand]; one block per token.
// ---------------------------------------------------------------------------
__global__ __launch_bounds__(256) void k_embed_add(
    float* __restrict__ X, const int* __restrict__ pos,
    const int* __restrict__ strand, const float* __restrict__ pe,
    const float* __restrict__ pos_tab, const float* __restrict__ strand_tab)
{
  long row = blockIdx.x;
  int p = pos[row]; p = p < 0 ? 0 : (p > 511 ? 511 : p);
  int s = strand[row];
  for (int c = threadIdx.x; c < HID; c += 256)
    X[row * HID + c] += pe[p * HID + c] + pos_tab[p * HID + c] + strand_tab[s * HID + c];
}

// ---------------------------------------------------------------------------
// Fused flash-attention with distance-bin + order bias.
// grid = (N/64 qtiles, NH, B), block = 128 (4 waves, 16 q-rows each).
// Q pre-scaled by 1/sqrt(DH); V staged transposed; output written as bf16.
// ---------------------------------------------------------------------------
#define LQ 72   // 144-B row stride: 16-B aligned, conflict-free

__global__ __launch_bounds__(128) void k_attn(
    const float* __restrict__ Qb, const float* __restrict__ Kb,
    const float* __restrict__ Vb, const int* __restrict__ positions,
    const float* __restrict__ mask, const float* __restrict__ distp,
    const float* __restrict__ orderp, __bf16* __restrict__ AO)
{
  __shared__ __attribute__((aligned(16))) __bf16 sQ[64][LQ];
  __shared__ __attribute__((aligned(16))) __bf16 sK[64][LQ];
  __shared__ __attribute__((aligned(16))) __bf16 sVt[64][LQ];   // [d][j]
  __shared__ __attribute__((aligned(16))) __bf16 sP[4][16][LQ];
  __shared__ int   posI[64], posJ[64];
  __shared__ float mI[64], mJ[64];

  int b = blockIdx.z, h = blockIdx.y, q0 = blockIdx.x * 64;
  int tid = threadIdx.x, wave = tid >> 5, lane = tid & 31;
  int nl = lane & 15, hh = lane >> 4;

  for (int q = tid; q < 64 * 16; q += 128) {
    int r = q >> 4, cq = (q & 15) << 2;
    float4 vv = *(const float4*)&Qb[((long)(b * NSEQ + q0 + r)) * HID + h * DH + cq];
    __bf16* dst = &sQ[r][cq];
    dst[0] = f2bf(vv.x * 0.125f); dst[1] = f2bf(vv.y * 0.125f);
    dst[2] = f2bf(vv.z * 0.125f); dst[3] = f2bf(vv.w * 0.125f);
  }
  for (int r = tid; r < 64; r += 128) {
    posI[r] = positions[b * NSEQ + q0 + r];
    mI[r]   = mask[b * NSEQ + q0 + r];
  }
  __syncthreads();

  v16bf aQ0 = load_frag_a(&sQ[wave * 16][0],  LQ);
  v16bf aQ1 = load_frag_a(&sQ[wave * 16][32], LQ);

  float m_run[8], l_run[8];
  v8f acc[4] = {};
#pragma unroll
  for (int r = 0; r < 8; ++r) { m_run[r] = -3.0e38f; l_run[r] = 0.0f; }
  float ord = orderp[h];

  for (int j0 = 0; j0 < NSEQ; j0 += 64) {
    for (int q = tid; q < 64 * 16; q += 128) {
      int r = q >> 4, cq = (q & 15) << 2;
      long g = ((long)(b * NSEQ + j0 + r)) * HID + h * DH + cq;
      float4 kv = *(const float4*)&Kb[g];
      __bf16* dk = &sK[r][cq];
      dk[0] = f2bf(kv.x); dk[1] = f2bf(kv.y); dk[2] = f2bf(kv.z); dk[3] = f2bf(kv.w);
      float4 vv = *(const float4*)&Vb[g];
      sVt[cq + 0][r] = f2bf(vv.x);
      sVt[cq + 1][r] = f2bf(vv.y);
      sVt[cq + 2][r] = f2bf(vv.z);
      sVt[cq + 3][r] = f2bf(vv.w);
    }
    for (int r = tid; r < 64; r += 128) {
      posJ[r] = positions[b * NSEQ + j0 + r];
      mJ[r]   = mask[b * NSEQ + j0 + r];
    }
    __syncthreads();

    // S = Q * K^T : Bt[j][d] == natural row-major K tile
    v8f sc[4];
#pragma unroll
    for (int t = 0; t < 4; ++t) {
      v8f s = {};
      v16bf b0 = load_frag_bt(&sK[t * 16][0],  LQ);
      s = __builtin_amdgcn_wmma_f32_16x16x32_bf16(false, aQ0, false, b0, (short)0, s, false, false);
      v16bf b1 = load_frag_bt(&sK[t * 16][32], LQ);
      s = __builtin_amdgcn_wmma_f32_16x16x32_bf16(false, aQ1, false, b1, (short)0, s, false, false);
      sc[t] = s;
    }

    // dist-bin bias (edges = logspace(0,3,31) => bin = ceil(10*log10 d)),
    // order term, mask
#pragma unroll
    for (int t = 0; t < 4; ++t) {
      int jloc = t * 16 + nl;
      int pj = posJ[jloc];
      float mj = mJ[jloc];
#pragma unroll
      for (int r = 0; r < 8; ++r) {
        int iloc = wave * 16 + (hh << 3) + r;
        int pi = posI[iloc];
        float v = sc[t][r];
        float d = fabsf((float)(pj - pi));
        int bin = 0;
        if (d >= 1.0f) {
          bin = (int)ceilf(3.0102999566f * __log2f(d));   // 10*log10(d)
          bin = bin < 0 ? 0 : (bin > 31 ? 31 : bin);
        }
        v += distp[bin * NHEAD + h];
        v += ((pj > pi) ? 0.5f : -0.5f) * ord;
        if (mj == 0.0f || mI[iloc] == 0.0f) v = -1.0e9f;
        sc[t][r] = v;
      }
    }

    // online softmax (row stats in 16-lane halves of the C layout)
    float m_new[8];
#pragma unroll
    for (int r = 0; r < 8; ++r) {
      float mm = fmaxf(fmaxf(sc[0][r], sc[1][r]), fmaxf(sc[2][r], sc[3][r]));
      mm = fmaxf(mm, __shfl_xor(mm, 1));
      mm = fmaxf(mm, __shfl_xor(mm, 2));
      mm = fmaxf(mm, __shfl_xor(mm, 4));
      mm = fmaxf(mm, __shfl_xor(mm, 8));
      m_new[r] = fmaxf(m_run[r], mm);
    }
#pragma unroll
    for (int r = 0; r < 8; ++r) {
      float rs = 0.0f;
#pragma unroll
      for (int t = 0; t < 4; ++t) {
        float p = __expf(sc[t][r] - m_new[r]);
        rs += p;
        sP[wave][(hh << 3) + r][t * 16 + nl] = f2bf(p);
      }
      rs += __shfl_xor(rs, 1);
      rs += __shfl_xor(rs, 2);
      rs += __shfl_xor(rs, 4);
      rs += __shfl_xor(rs, 8);
      float alpha = __expf(m_run[r] - m_new[r]);
      l_run[r] = l_run[r] * alpha + rs;
      m_run[r] = m_new[r];
#pragma unroll
      for (int t = 0; t < 4; ++t) acc[t][r] *= alpha;
    }

    // O += P * V : Bt[d][j] == sVt
    v16bf aP0 = load_frag_a(&sP[wave][0][0],  LQ);
    v16bf aP1 = load_frag_a(&sP[wave][0][32], LQ);
#pragma unroll
    for (int t = 0; t < 4; ++t) {
      v16bf bv0 = load_frag_bt(&sVt[t * 16][0],  LQ);
      acc[t] = __builtin_amdgcn_wmma_f32_16x16x32_bf16(false, aP0, false, bv0, (short)0, acc[t], false, false);
      v16bf bv1 = load_frag_bt(&sVt[t * 16][32], LQ);
      acc[t] = __builtin_amdgcn_wmma_f32_16x16x32_bf16(false, aP1, false, bv1, (short)0, acc[t], false, false);
    }
    __syncthreads();
  }

#pragma unroll
  for (int r = 0; r < 8; ++r) {
    int iloc = wave * 16 + (hh << 3) + r;
    float inv = l_run[r] > 0.0f ? 1.0f / l_run[r] : 0.0f;
    float mrow = mI[iloc];
    long base = ((long)(b * NSEQ + q0 + iloc)) * HID + h * DH;
#pragma unroll
    for (int t = 0; t < 4; ++t)
      AO[base + t * 16 + nl] = f2bf(acc[t][r] * inv * mrow);
  }
}

// ---------------------------------------------------------------------------
// Pool query projection: pqv[n] = sum_k pool_q[k]*pq_w[k,n] + pq_b[n]
// ---------------------------------------------------------------------------
__global__ __launch_bounds__(256) void k_poolq(
    const float* __restrict__ pool_q, const float* __restrict__ pq_w,
    const float* __restrict__ pq_b, float* __restrict__ pqv)
{
  int n = blockIdx.x * 256 + threadIdx.x;
  if (n >= HID) return;
  float s = pq_b[n];
  for (int k = 0; k < HID; ++k) s += pool_q[k] * pq_w[(long)k * HID + n];
  pqv[n] = s;
}

// ---------------------------------------------------------------------------
// Pooled attention (1 query per (b,h)); grid=(NH,B), block=256. bf16 output.
// ---------------------------------------------------------------------------
__global__ __launch_bounds__(256) void k_pool_attn(
    const float* __restrict__ K2, const float* __restrict__ V2,
    const float* __restrict__ pqv, const float* __restrict__ mask,
    __bf16* __restrict__ pooled)
{
  __shared__ float sa[NSEQ];
  __shared__ float red[256];
  __shared__ float qh[DH];
  int h = blockIdx.x, b = blockIdx.y;
  int tid = threadIdx.x;

  if (tid < DH) qh[tid] = pqv[h * DH + tid] * 0.125f;
  __syncthreads();

  for (int j = tid; j < NSEQ; j += 256) {
    float s = 0.0f;
    long base = ((long)(b * NSEQ + j)) * HID + h * DH;
    for (int d = 0; d < DH; ++d) s += qh[d] * K2[base + d];
    sa[j] = (mask[b * NSEQ + j] == 0.0f) ? -3.0e38f : s;
  }
  __syncthreads();

  red[tid] = fmaxf(sa[tid], sa[tid + 256]);
  __syncthreads();
  for (int st = 128; st > 0; st >>= 1) {
    if (tid < st) red[tid] = fmaxf(red[tid], red[tid + st]);
    __syncthreads();
  }
  float m = red[0];
  __syncthreads();

  float e0 = __expf(sa[tid] - m), e1 = __expf(sa[tid + 256] - m);
  sa[tid] = e0; sa[tid + 256] = e1;
  red[tid] = e0 + e1;
  __syncthreads();
  for (int st = 128; st > 0; st >>= 1) {
    if (tid < st) red[tid] += red[tid + st];
    __syncthreads();
  }
  float S = red[0];
  float inv = (m > -1.0e38f && S > 0.0f) ? 1.0f / S : 0.0f;  // nan_to_num guard
  __syncthreads();

  if (tid < DH) {
    float o = 0.0f;
    for (int j = 0; j < NSEQ; ++j)
      o += sa[j] * V2[((long)(b * NSEQ + j)) * HID + h * DH + tid];
    pooled[(long)b * HID + h * DH + tid] = f2bf(o * inv);
  }
}

// ---------------------------------------------------------------------------
// Host orchestration
// ---------------------------------------------------------------------------
static inline dim3 gemm_grid(int M, int N) {
  return dim3((unsigned)((N + TN - 1) / TN), (unsigned)((M + TM - 1) / TM), 1);
}

extern "C" void kernel_launch(void* const* d_in, const int* in_sizes, int n_in,
                              void* d_out, int out_size, void* d_ws, size_t ws_size,
                              hipStream_t stream)
{
  (void)in_sizes; (void)n_in; (void)out_size; (void)ws_size;

  // inputs (setup_inputs dict order; params pytree-flattened with sorted keys)
  const float* emb    = (const float*)d_in[0];
  const int*   pos    = (const int*)  d_in[1];
  const int*   strand = (const int*)  d_in[2];
  const float* mask   = (const float*)d_in[3];
  const float* in_b   = (const float*)d_in[4];
  const float* in_w   = (const float*)d_in[5];
  const float* op1_b  = (const float*)d_in[60];
  const float* op1_w  = (const float*)d_in[61];
  const float* op2_b  = (const float*)d_in[62];
  const float* op2_w  = (const float*)d_in[63];
  const float* pe     = (const float*)d_in[64];
  const float* pk_b   = (const float*)d_in[65];
  const float* pk_w   = (const float*)d_in[66];
  const float* po_b   = (const float*)d_in[67];
  const float* po_w   = (const float*)d_in[68];
  const float* pool_q = (const float*)d_in[69];
  const float* pos_tab= (const float*)d_in[70];
  const float* pq_b   = (const float*)d_in[71];
  const float* pq_w   = (const float*)d_in[72];
  const float* pv_b   = (const float*)d_in[73];
  const float* pv_w   = (const float*)d_in[74];
  const float* strand_tab = (const float*)d_in[75];

  // ---------------- workspace layout ----------------
  const long RC = (long)MROWS * HID;   // 4,194,304
  char* wp = (char*)d_ws;
  auto allocF = [&](long n) { float* p = (float*)wp; wp += n * sizeof(float); return p; };
  auto allocB = [&](long n) { __bf16* p = (__bf16*)wp; wp += n * sizeof(__bf16); return p; };

  float*  X   = allocF(RC);                 // f32 residual stream
  float*  Qf  = allocF(RC);
  float*  Kf  = allocF(RC);
  float*  Vf  = allocF(RC);
  float*  PQV = allocF(HID);
  __bf16* Eb  = allocB((long)MROWS * 768);  // emb bf16
  __bf16* Tb  = allocB(RC);                 // LN output
  __bf16* AOb = allocB(RC);                 // attention output
  __bf16* Hhb = allocB((long)MROWS * 2048); // FFN hidden
  __bf16* Xb  = allocB(RC);                 // X bf16 (pool path)
  __bf16* POOLb = allocB((long)NBATCH * HID);
  __bf16* P2b   = allocB((long)NBATCH * HID);
  __bf16* OP1b  = allocB((long)NBATCH * HID);
  // transposed bf16 weights
  __bf16* inw_t = allocB(768L * HID);
  __bf16* lw_t[3][6];   // q,k,v,o,f1,f2
  for (int l = 0; l < 3; ++l) {
    lw_t[l][0] = allocB((long)HID * HID);
    lw_t[l][1] = allocB((long)HID * HID);
    lw_t[l][2] = allocB((long)HID * HID);
    lw_t[l][3] = allocB((long)HID * HID);
    lw_t[l][4] = allocB((long)HID * 2048);
    lw_t[l][5] = allocB(2048L * HID);
  }
  __bf16* pkw_t  = allocB((long)HID * HID);
  __bf16* pvw_t  = allocB((long)HID * HID);
  __bf16* pow_t  = allocB((long)HID * HID);
  __bf16* op1w_t = allocB((long)HID * HID);
  __bf16* op2w_t = allocB((long)HID * 128);

  auto wtr = [&](const float* W, __bf16* Wt, int K, int N) {
    long n = (long)K * N;
    k_wtr<<<dim3((unsigned)((n + 255) / 256)), 256, 0, stream>>>(W, Wt, K, N);
  };

  // 0) convert operands / transpose weights to bf16
  k_cvt<<<dim3((unsigned)((long)MROWS * 768 / 4 / 256)), 256, 0, stream>>>(emb, Eb, (long)MROWS * 768);
  wtr(in_w, inw_t, 768, HID);
  for (int l = 0; l < 3; ++l) {
    int base = 6 + 18 * l;
    wtr((const float*)d_in[base + 15], lw_t[l][0], HID, HID);   // q_w
    wtr((const float*)d_in[base + 6],  lw_t[l][1], HID, HID);   // k_w
    wtr((const float*)d_in[base + 17], lw_t[l][2], HID, HID);   // v_w
    wtr((const float*)d_in[base + 12], lw_t[l][3], HID, HID);   // o_w
    wtr((const float*)d_in[base + 2],  lw_t[l][4], HID, 2048);  // f1_w
    wtr((const float*)d_in[base + 4],  lw_t[l][5], 2048, HID);  // f2_w
  }
  wtr(pk_w,  pkw_t,  HID, HID);
  wtr(pv_w,  pvw_t,  HID, HID);
  wtr(po_w,  pow_t,  HID, HID);
  wtr(op1_w, op1w_t, HID, HID);
  wtr(op2_w, op2w_t, HID, 128);

  // 1) X = emb @ in_w + in_b ; then table adds
  k_gemm<<<gemm_grid(MROWS, HID), 256, 0, stream>>>(Eb, inw_t, in_b, nullptr, X, nullptr,
                                                    MROWS, HID, 768, 0);
  k_embed_add<<<MROWS, 256, 0, stream>>>(X, pos, strand, pe, pos_tab, strand_tab);

  // 2) transformer layers
  for (int l = 0; l < 3; ++l) {
    int base = 6 + 18 * l;   // sorted leaf order inside each layer dict
    const float* dist = (const float*)d_in[base + 0];
    const float* f1_b = (const float*)d_in[base + 1];
    const float* f2_b = (const float*)d_in[base + 3];
    const float* k_b  = (const float*)d_in[base + 5];
    const float* n1_b = (const float*)d_in[base + 7];
    const float* n1_g = (const float*)d_in[base + 8];
    const float* n2_b = (const float*)d_in[base + 9];
    const float* n2_g = (const float*)d_in[base + 10];
    const float* o_b  = (const float*)d_in[base + 11];
    const float* ordp = (const float*)d_in[base + 13];
    const float* q_b  = (const float*)d_in[base + 14];
    const float* v_b  = (const float*)d_in[base + 16];

    k_layernorm<<<MROWS, 256, 0, stream>>>(X, n1_g, n1_b, Tb, HID);
    k_gemm<<<gemm_grid(MROWS, HID), 256, 0, stream>>>(Tb, lw_t[l][0], q_b, nullptr, Qf, nullptr, MROWS, HID, HID, 0);
    k_gemm<<<gemm_grid(MROWS, HID), 256, 0, stream>>>(Tb, lw_t[l][1], k_b, nullptr, Kf, nullptr, MROWS, HID, HID, 0);
    k_gemm<<<gemm_grid(MROWS, HID), 256, 0, stream>>>(Tb, lw_t[l][2], v_b, nullptr, Vf, nullptr, MROWS, HID, HID, 0);
    k_attn<<<dim3(NSEQ / 64, NHEAD, NBATCH), 128, 0, stream>>>(Qf, Kf, Vf, pos, mask,
                                                               dist, ordp, AOb);
    // x = x + AO @ o_w + o_b  (in-place residual)
    k_gemm<<<gemm_grid(MROWS, HID), 256, 0, stream>>>(AOb, lw_t[l][3], o_b, X, X, nullptr, MROWS, HID, HID, 0);
    // FFN
    k_layernorm<<<MROWS, 256, 0, stream>>>(X, n2_g, n2_b, Tb, HID);
    k_gemm<<<gemm_grid(MROWS, 2048), 256, 0, stream>>>(Tb, lw_t[l][4], f1_b, nullptr, nullptr, Hhb,
                                                       MROWS, 2048, HID, 1 /*gelu*/);
    k_gemm<<<gemm_grid(MROWS, HID), 256, 0, stream>>>(Hhb, lw_t[l][5], f2_b, X, X, nullptr,
                                                      MROWS, HID, 2048, 0);
  }

  // 3) pooled attention head
  k_cvt<<<dim3((unsigned)(RC / 4 / 256)), 256, 0, stream>>>(X, Xb, RC);
  k_gemm<<<gemm_grid(MROWS, HID), 256, 0, stream>>>(Xb, pkw_t, pk_b, nullptr, Qf, nullptr, MROWS, HID, HID, 0); // K2
  k_gemm<<<gemm_grid(MROWS, HID), 256, 0, stream>>>(Xb, pvw_t, pv_b, nullptr, Kf, nullptr, MROWS, HID, HID, 0); // V2
  k_poolq<<<2, 256, 0, stream>>>(pool_q, pq_w, pq_b, PQV);
  k_pool_attn<<<dim3(NHEAD, NBATCH), 256, 0, stream>>>(Qf, Kf, PQV, mask, POOLb);
  // o @ po_w + po_b
  k_gemm<<<gemm_grid(NBATCH, HID), 256, 0, stream>>>(POOLb, pow_t, po_b, nullptr, nullptr, P2b,
                                                     NBATCH, HID, HID, 0);
  // h = gelu(pooled @ op1_w + op1_b)
  k_gemm<<<gemm_grid(NBATCH, HID), 256, 0, stream>>>(P2b, op1w_t, op1_b, nullptr, nullptr, OP1b,
                                                     NBATCH, HID, HID, 1);
  // out = h @ op2_w + op2_b
  k_gemm<<<gemm_grid(NBATCH, 128), 256, 0, stream>>>(OP1b, op2w_t, op2_b, nullptr,
                                                     (float*)d_out, nullptr, NBATCH, 128, HID, 0);
}
